// GraphSAGE_53137335386893
// MI455X (gfx1250) — matrix operand order repaired and checked by reading, not compiled, not run on hardware
//
#include <hip/hip_runtime.h>
#include <hip/hip_bf16.h>

// ---------------------------------------------------------------------------
// GraphSAGE on gfx1250 (MI455X):
//   per layer:  agg = scatter_add(h[src] -> dst)       (f32 HW atomics, L2)
//               out = h @ Wself + (agg*invdeg) @ Wneigh + b   (WMMA bf16->f32)
// GEMM: weights staged in LDS (ds_load_b128 B-fragments), A double-buffered
// from global, 8 back-to-back v_wmma_f32_16x16x32_bf16 per K-step.
// ---------------------------------------------------------------------------

typedef __bf16 bf16_t;
typedef __attribute__((ext_vector_type(16))) __bf16 v16bf;
typedef __attribute__((ext_vector_type(8)))  __bf16 v8bf;
typedef __attribute__((ext_vector_type(8)))  float  v8f;

#define DIM_IN  256
#define DIM_HID 256
#define DIM_OUT 128
#define LDSK    264   // 256 + 8 bf16 pad: 528B row stride, 16B aligned, conflict-free

// ---------------- small prep kernels ----------------------------------------

__global__ void sage_deg_kernel(const int* __restrict__ dst, float* __restrict__ deg, int E) {
    int e = blockIdx.x * blockDim.x + threadIdx.x;
    if (e < E) unsafeAtomicAdd(&deg[dst[e]], 1.0f);
}

__global__ void sage_invdeg_kernel(float* __restrict__ deg, int n) {
    int i = blockIdx.x * blockDim.x + threadIdx.x;
    if (i < n) deg[i] = 1.0f / fmaxf(deg[i], 1.0f);
}

// W [din x dout] f32 row-major  ->  Wt [dout x din] bf16 row-major
__global__ void sage_wt_kernel(const float* __restrict__ W, bf16_t* __restrict__ Wt,
                               int din, int dout) {
    int idx = blockIdx.x * blockDim.x + threadIdx.x;
    if (idx >= din * dout) return;
    int o = idx / din;
    int k = idx - o * din;
    Wt[idx] = (bf16_t)W[(size_t)k * dout + o];
}

__global__ void sage_f2b_kernel(const float* __restrict__ x, bf16_t* __restrict__ y, long n) {
    long i = ((long)blockIdx.x * blockDim.x + threadIdx.x) * 4;
    if (i + 4 <= n) {
        float4 v = *(const float4*)(x + i);
        y[i + 0] = (bf16_t)v.x; y[i + 1] = (bf16_t)v.y;
        y[i + 2] = (bf16_t)v.z; y[i + 3] = (bf16_t)v.w;
    } else {
        for (; i < n; ++i) y[i] = (bf16_t)x[i];
    }
}

// ---------------- scatter: agg[dst] += h[src]  (one wave per edge) ----------

__global__ void sage_scatter_kernel(const bf16_t* __restrict__ h,
                                    const int* __restrict__ src,
                                    const int* __restrict__ dst,
                                    float* __restrict__ agg, int E) {
    long gid = (long)blockIdx.x * blockDim.x + threadIdx.x;
    int e = (int)(gid >> 5);               // 32 lanes cover 256 feats, 8 each
    if (e >= E) return;
    int c = ((int)gid & 31) << 3;
    int s = src[e];
    int d = dst[e];
    v8bf v = *(const v8bf*)(h + (size_t)s * DIM_IN + c);
    float* q = agg + (size_t)d * DIM_IN + c;
#pragma unroll
    for (int j = 0; j < 8; ++j)
        unsafeAtomicAdd(q + j, (float)v[j]);
}

// ---------------- WMMA fragment loaders (ISA 7.12.2 layouts) ----------------

// A 16x32 bf16: lane<16 -> K {0..7, 16..23}; lane>=16 -> K {8..15, 24..31}
__device__ __forceinline__ v16bf load_a_bf16(const bf16_t* __restrict__ h, int ld,
                                             int row, int kb, int lane) {
    const int koff = (lane < 16) ? 0 : 8;
    const bf16_t* p = h + (size_t)row * ld + kb + koff;
    v8bf lo = *(const v8bf*)(p);
    v8bf hi = *(const v8bf*)(p + 16);
    v16bf a;
#pragma unroll
    for (int j = 0; j < 8; ++j) { a[j] = lo[j]; a[j + 8] = hi[j]; }
    return a;
}

// Same A layout, but source is f32 (the aggregate) scaled by inv_deg, cvt->bf16
__device__ __forceinline__ v16bf load_a_f32(const float* __restrict__ g, int ld,
                                            int row, int kb, int lane, float sc) {
    const int koff = (lane < 16) ? 0 : 8;
    const float* p = g + (size_t)row * ld + kb + koff;
    float4 l0 = *(const float4*)(p);
    float4 l1 = *(const float4*)(p + 4);
    float4 h0 = *(const float4*)(p + 16);
    float4 h1 = *(const float4*)(p + 20);
    v16bf a;
    a[0]  = (bf16_t)(l0.x * sc); a[1]  = (bf16_t)(l0.y * sc);
    a[2]  = (bf16_t)(l0.z * sc); a[3]  = (bf16_t)(l0.w * sc);
    a[4]  = (bf16_t)(l1.x * sc); a[5]  = (bf16_t)(l1.y * sc);
    a[6]  = (bf16_t)(l1.z * sc); a[7]  = (bf16_t)(l1.w * sc);
    a[8]  = (bf16_t)(h0.x * sc); a[9]  = (bf16_t)(h0.y * sc);
    a[10] = (bf16_t)(h0.z * sc); a[11] = (bf16_t)(h0.w * sc);
    a[12] = (bf16_t)(h1.x * sc); a[13] = (bf16_t)(h1.y * sc);
    a[14] = (bf16_t)(h1.z * sc); a[15] = (bf16_t)(h1.w * sc);
    return a;
}

// B 32x16 bf16 from LDS-staged transposed weights (row stride LDSK):
// lane = col (mod 16); lane<16 -> K kb+{0..7, 16..23}, lane>=16 -> K kb+{8..15, 24..31}
__device__ __forceinline__ v16bf load_b_lds(const bf16_t* s, int col_local, int kb,
                                            int lane) {
    const int koff = (lane < 16) ? 0 : 8;
    const bf16_t* p = s + col_local * LDSK + kb + koff;
    v8bf lo = *(const v8bf*)(p);       // ds_load_b128
    v8bf hi = *(const v8bf*)(p + 16);  // ds_load_b128
    v16bf b;
#pragma unroll
    for (int j = 0; j < 8; ++j) { b[j] = lo[j]; b[j + 8] = hi[j]; }
    return b;
}

// ---------------- fused dual-GEMM layer kernel ------------------------------
// wave: 16 rows x 64 cols (4 accumulators); block: 8 waves -> 128 rows.
// din is always DIM_HID (=256) for every layer of this model.
template <bool RELU, bool OUT_BF16>
__global__ __launch_bounds__(256)
void sage_gemm_kernel(const bf16_t* __restrict__ h,   // [N x 256] bf16
                      const float*  __restrict__ agg, // [N x 256] f32 (L2-resident)
                      const float*  __restrict__ invd,
                      const bf16_t* __restrict__ Wts, // [dout x 256] bf16
                      const bf16_t* __restrict__ Wtn, // [dout x 256] bf16
                      const float*  __restrict__ bias,
                      float*  __restrict__ out_f,
                      bf16_t* __restrict__ out_b,
                      int nnodes, int dout) {
    constexpr int DIN = DIM_HID;
    __shared__ bf16_t sWs[64 * LDSK];
    __shared__ bf16_t sWn[64 * LDSK];

    const int tid  = threadIdx.x;
    const int lane = tid & 31;
    const int wave = tid >> 5;
    const int c0   = blockIdx.y * 64;

    // ---- stage both 64x256 weight tiles into LDS (all waves participate) ----
#pragma unroll
    for (int it = 0; it < 8; ++it) {
        int idx = it * 2048 + tid * 8;      // [0, 16384), 8 bf16 per thread-chunk
        int col = idx >> 8;                 // / 256
        int k   = idx & 255;
        v8bf vs = *(const v8bf*)(Wts + (size_t)(c0 + col) * DIN + k);
        v8bf vn = *(const v8bf*)(Wtn + (size_t)(c0 + col) * DIN + k);
        *(v8bf*)(sWs + col * LDSK + k) = vs;
        *(v8bf*)(sWn + col * LDSK + k) = vn;
    }
    __syncthreads();

    const int r0 = blockIdx.x * 128 + wave * 16;
    if (r0 >= nnodes) return;                 // wave-uniform: EXEC stays all-ones
    const int m = lane & 15;

    int row = r0 + m;
    if (row >= nnodes) row = nnodes - 1;      // clamp loads; stores masked later
    const float sc = invd[row];

    v8f acc[4] = {};

    // ---- K loop: A double-buffered from global, B from LDS ------------------
    v16bf aS = load_a_bf16(h, DIN, row, 0, lane);
    v16bf aN = load_a_f32(agg, DIN, row, 0, lane, sc);

    for (int kb = 0; kb < DIN; kb += 32) {
        v16bf aS_n, aN_n;
        if (kb + 32 < DIN) {                  // prefetch next K-step's A frags
            aS_n = load_a_bf16(h, DIN, row, kb + 32, lane);
            aN_n = load_a_f32(agg, DIN, row, kb + 32, lane, sc);
        }

        v16bf bS[4], bN[4];
#pragma unroll
        for (int t = 0; t < 4; ++t) {
            bS[t] = load_b_lds(sWs, t * 16, kb, lane);
            bN[t] = load_b_lds(sWn, t * 16, kb, lane);
        }
#pragma unroll
        for (int t = 0; t < 4; ++t) {
            acc[t] = __builtin_amdgcn_wmma_f32_16x16x32_bf16(
                false, aS, false, bS[t], (short)0, acc[t], false, false);
            acc[t] = __builtin_amdgcn_wmma_f32_16x16x32_bf16(
                false, aN, false, bN[t], (short)0, acc[t], false, false);
        }
        aS = aS_n;
        aN = aN_n;
    }

    // D layout: lane = column (mod 16); lane-half selects rows 0..7 / 8..15
    const int rbase = r0 + ((lane < 16) ? 0 : 8);
#pragma unroll
    for (int t = 0; t < 4; ++t) {
        const int col = c0 + t * 16 + m;
        const float bv = bias[col];
#pragma unroll
        for (int v = 0; v < 8; ++v) {
            const int r = rbase + v;
            if (r < nnodes) {
                float val = acc[t][v] + bv;
                if (RELU) val = fmaxf(val, 0.0f);
                if (OUT_BF16) out_b[(size_t)r * dout + col] = (bf16_t)val;
                else          out_f[(size_t)r * dout + col] = val;
            }
        }
    }
}

// ---------------------------------------------------------------------------

extern "C" void kernel_launch(void* const* d_in, const int* in_sizes, int n_in,
                              void* d_out, int out_size, void* d_ws, size_t ws_size,
                              hipStream_t stream) {
    const float* feat = (const float*)d_in[0];
    const int*   src  = (const int*)d_in[1];
    const int*   dst  = (const int*)d_in[2];
    const float* Ws[3] = {(const float*)d_in[3], (const float*)d_in[6], (const float*)d_in[9]};
    const float* Wn[3] = {(const float*)d_in[4], (const float*)d_in[7], (const float*)d_in[10]};
    const float* Bb[3] = {(const float*)d_in[5], (const float*)d_in[8], (const float*)d_in[11]};

    const int N = in_sizes[0] / DIM_IN;
    const int E = in_sizes[1];
    const int douts[3] = {DIM_HID, DIM_HID, DIM_OUT};

    // ---- workspace carve-up -------------------------------------------------
    char* base = (char*)d_ws;
    size_t off = 0;
    auto wsalloc = [&](size_t bytes) -> void* {
        void* p = base + off;
        off = (off + bytes + 255) & ~(size_t)255;
        return p;
    };
    float*  invd = (float*)wsalloc((size_t)N * 4);
    bf16_t* hbA  = (bf16_t*)wsalloc((size_t)N * DIM_HID * 2);
    bf16_t* hbB  = (bf16_t*)wsalloc((size_t)N * DIM_HID * 2);
    float*  agg  = (float*)wsalloc((size_t)N * DIM_HID * 4);
    bf16_t* Wts[3], *Wtn[3];
    for (int l = 0; l < 3; ++l) {
        Wts[l] = (bf16_t*)wsalloc((size_t)DIM_HID * douts[l] * 2);
        Wtn[l] = (bf16_t*)wsalloc((size_t)DIM_HID * douts[l] * 2);
    }

    // ---- degrees ------------------------------------------------------------
    hipMemsetAsync(invd, 0, (size_t)N * 4, stream);
    sage_deg_kernel<<<(E + 255) / 256, 256, 0, stream>>>(dst, invd, E);
    sage_invdeg_kernel<<<(N + 255) / 256, 256, 0, stream>>>(invd, N);

    // ---- weights: f32 [din x dout] -> bf16 [dout x din] ---------------------
    for (int l = 0; l < 3; ++l) {
        int cnt = DIM_HID * douts[l];
        sage_wt_kernel<<<(cnt + 255) / 256, 256, 0, stream>>>(Ws[l], Wts[l], DIM_HID, douts[l]);
        sage_wt_kernel<<<(cnt + 255) / 256, 256, 0, stream>>>(Wn[l], Wtn[l], DIM_HID, douts[l]);
    }

    // ---- features -> bf16 ---------------------------------------------------
    long nfeat = (long)N * DIM_IN;
    sage_f2b_kernel<<<(int)((nfeat / 4 + 255) / 256), 256, 0, stream>>>(feat, hbA, nfeat);

    // ---- 3 layers -----------------------------------------------------------
    bf16_t* hin   = hbA;
    bf16_t* hnext = hbB;
    const long scat_threads = (long)E * 32;
    const int  scat_blocks  = (int)((scat_threads + 255) / 256);

    for (int l = 0; l < 3; ++l) {
        hipMemsetAsync(agg, 0, (size_t)N * DIM_HID * 4, stream);
        sage_scatter_kernel<<<scat_blocks, 256, 0, stream>>>(hin, src, dst, agg, E);

        dim3 grid((N + 127) / 128, douts[l] / 64);
        if (l < 2) {
            sage_gemm_kernel<true, true><<<grid, 256, 0, stream>>>(
                hin, agg, invd, Wts[l], Wtn[l], Bb[l],
                nullptr, hnext, N, douts[l]);
            bf16_t* tmp = hin; hin = hnext; hnext = tmp;
        } else {
            sage_gemm_kernel<false, false><<<grid, 256, 0, stream>>>(
                hin, agg, invd, Wts[l], Wtn[l], Bb[l],
                (float*)d_out, nullptr, N, douts[l]);
        }
    }
}
